// ASLayer_14250701488903
// MI455X (gfx1250) — matrix operand order repaired and checked by reading, not compiled, not run on hardware
//
#include <hip/hip_runtime.h>
#include <stdint.h>

#define N_NODES 50000
#define N_EDGES 800000
#define DIN     512
#define DOUT    256
#define STRIPES (N_NODES / 16)   // 3125, exact
#define SLICE_DW 4096            // one kt-slice of packed W: 16 nt * 32 lanes * 8 dw = 16KB

typedef __attribute__((ext_vector_type(16))) __bf16 v16bf;
typedef __attribute__((ext_vector_type(8)))  float  v8f;

union BPack { uint32_t u[8]; uint4 q[2]; v16bf v; };

__device__ __forceinline__ uint32_t pack_bf16(float lo, float hi) {
    uint32_t a = __float_as_uint(lo), b = __float_as_uint(hi);
    a = (a + 0x7FFFu + ((a >> 16) & 1u)) >> 16;   // round-to-nearest-even
    b = (b + 0x7FFFu + ((b >> 16) & 1u)) >> 16;
    return a | (b << 16);
}

// ---------------------------------------------------------------------------
// Pre-pack W (512x256 row-major f32) into the WMMA B-operand layout, bf16.
// Per (kt, nt) tile of 32x16: lane holds col n = nt*16+(lane&15); lanes 0-15
// carry K = kt*32 + 0..15 (2 per dword), lanes 16-31 carry K = kt*32 + 16..31.
// Index = ((kt*16+nt)*32 + lane)*8 + v  -> contiguous 8 dwords per lane.
// ---------------------------------------------------------------------------
__global__ void pack_w_kernel(const float* __restrict__ W, uint32_t* __restrict__ Wbf) {
    int idx  = blockIdx.x * 256 + threadIdx.x;       // 65536 total
    int v    = idx & 7;
    int lane = (idx >> 3) & 31;
    int nt   = (idx >> 8) & 15;
    int kt   = idx >> 12;
    int n    = nt * 16 + (lane & 15);
    int k0   = kt * 32 + (lane >> 4) * 16 + 2 * v;
    Wbf[idx] = pack_bf16(W[k0 * DOUT + n], W[(k0 + 1) * DOUT + n]);
}

// ---------------------------------------------------------------------------
// el/er: fused dual GEMV, one wave (32 lanes) per node, shfl_xor reduction.
// ---------------------------------------------------------------------------
__global__ __launch_bounds__(256)
void node_attn_kernel(const float* __restrict__ feat,
                      const float* __restrict__ Wl, const float* __restrict__ Wr,
                      const float* __restrict__ bl, const float* __restrict__ br,
                      float* __restrict__ el, float* __restrict__ er) {
    int lane = threadIdx.x & 31;
    int node = blockIdx.x * 8 + (threadIdx.x >> 5);
    if (node >= N_NODES) return;
    const float* row = feat + (size_t)node * DIN;
    float sl = 0.f, sr = 0.f;
    #pragma unroll
    for (int k = 0; k < DIN; k += 32) {
        float f = row[k + lane];
        sl = fmaf(f, Wl[k + lane], sl);
        sr = fmaf(f, Wr[k + lane], sr);
    }
    #pragma unroll
    for (int off = 16; off > 0; off >>= 1) {
        sl += __shfl_xor(sl, off, 32);
        sr += __shfl_xor(sr, off, 32);
    }
    if (lane == 0) { el[node] = sl + bl[0]; er[node] = sr + br[0]; }
}

// ---------------------------------------------------------------------------
// ft = feat @ W + b via v_wmma_f32_16x16x32_bf16.
// 8 waves/block, one 16-row stripe per wave. The B panel (one kt-slice,
// 16KB) is shared by all waves: double-buffered in LDS, filled with
// global_load_async_to_lds_b128 (ASYNCcnt) overlapping the kt compute.
// B-operand ds_loads are software-pipelined depth-2 through 3 register
// buffers so each WMMA never waits on a just-issued LDS read.
// ---------------------------------------------------------------------------
__global__ __launch_bounds__(256)
void gemm_ft_kernel(const float* __restrict__ feat,
                    const uint32_t* __restrict__ Wbf,
                    const float* __restrict__ bias,
                    float* __restrict__ ft) {
    __shared__ uint32_t bsh[2 * SLICE_DW];           // 32KB of the 320KB WGP LDS

    int tid    = threadIdx.x;
    int lane   = tid & 31;
    int stripe = blockIdx.x * 8 + (tid >> 5);        // wave-uniform
    if (stripe >= STRIPES) stripe = STRIPES - 1;     // clamp: keep all waves in
                                                     // barriers, duplicate stores
                                                     // write identical values
    int hi   = lane >> 4;
    int rowA = stripe * 16 + (lane & 15);
    const float* arow = feat + (size_t)rowA * DIN;

    // Each thread copies 64B (4 x b128) of a 16KB kt-slice into LDS.
    // ISA: INST_OFFSET is added to BOTH the LDS and global addresses.
    auto load_slice = [&](int kt, int sel) {
        const uint32_t* g   = Wbf + kt * SLICE_DW + tid * 16;
        uint32_t lds0 = (uint32_t)(uintptr_t)&bsh[sel * SLICE_DW + tid * 16];
        asm volatile("global_load_async_to_lds_b128 %0, %1, off"
                     :: "v"(lds0), "v"(g) : "memory");
        asm volatile("global_load_async_to_lds_b128 %0, %1, off offset:16"
                     :: "v"(lds0), "v"(g) : "memory");
        asm volatile("global_load_async_to_lds_b128 %0, %1, off offset:32"
                     :: "v"(lds0), "v"(g) : "memory");
        asm volatile("global_load_async_to_lds_b128 %0, %1, off offset:48"
                     :: "v"(lds0), "v"(g) : "memory");
    };

    v8f acc[16];
    #pragma unroll
    for (int nt = 0; nt < 16; ++nt) {
        float bv = bias[nt * 16 + (lane & 15)];
        #pragma unroll
        for (int r = 0; r < 8; ++r) acc[nt][r] = bv;
    }

    load_slice(0, 0);
    asm volatile("s_wait_asynccnt 0x0" ::: "memory");
    __syncthreads();

    for (int kt = 0; kt < 16; ++kt) {
        int cur = kt & 1;
        if (kt + 1 < 16) load_slice(kt + 1, cur ^ 1);   // prefetch next panel

        // A operand, contiguous k order (same as packed W): lanes 0-15 hold K = 0..15,
        // lanes 16-31 hold K = 16..31 (within this kt block).
        int kb = kt * 32 + hi * 16;
        float4 f0 = *(const float4*)(arow + kb);
        float4 f1 = *(const float4*)(arow + kb + 4);
        float4 f2 = *(const float4*)(arow + kb + 8);
        float4 f3 = *(const float4*)(arow + kb + 12);
        BPack a;
        a.u[0] = pack_bf16(f0.x, f0.y);  a.u[1] = pack_bf16(f0.z, f0.w);
        a.u[2] = pack_bf16(f1.x, f1.y);  a.u[3] = pack_bf16(f1.z, f1.w);
        a.u[4] = pack_bf16(f2.x, f2.y);  a.u[5] = pack_bf16(f2.z, f2.w);
        a.u[6] = pack_bf16(f3.x, f3.y);  a.u[7] = pack_bf16(f3.z, f3.w);

        const uint32_t* bbase = &bsh[cur * SLICE_DW + lane * 8];
        auto loadB = [&](BPack& b, int nt) {
            const uint32_t* bp = bbase + nt * 256;      // (nt*32 + lane)*8
            b.q[0] = *(const uint4*)bp;                 // ds_load_b128
            b.q[1] = *(const uint4*)(bp + 4);           // ds_load_b128
        };

        // Depth-2 software pipeline over 3 register buffers: the ds_loads for
        // tile nt are in flight while WMMAs for nt-2 and nt-1 execute.
        BPack b0, b1, b2;
        loadB(b0, 0);
        loadB(b1, 1);
        #pragma unroll
        for (int nt = 0; nt < 16; ++nt) {
            if (nt + 2 < 16) {
                BPack& bn = ((nt + 2) % 3 == 0) ? b0 : ((nt + 2) % 3 == 1) ? b1 : b2;
                loadB(bn, nt + 2);
            }
            BPack& bc = (nt % 3 == 0) ? b0 : (nt % 3 == 1) ? b1 : b2;
            acc[nt] = __builtin_amdgcn_wmma_f32_16x16x32_bf16(
                false, a.v, false, bc.v, (short)0, acc[nt], false, false);
        }

        if (kt + 1 < 16)
            asm volatile("s_wait_asynccnt 0x0" ::: "memory");
        __syncthreads();   // next panel visible to all waves; closes WAR on buffer
    }

    // C/D layout: lane (0-15|16-31) holds N = lane&15, VGPR r -> M = r | r+8.
    int rowC = stripe * 16 + hi * 8;
    int col0 = lane & 15;
    #pragma unroll
    for (int nt = 0; nt < 16; ++nt) {
        #pragma unroll
        for (int r = 0; r < 8; ++r)
            ft[(size_t)(rowC + r) * DOUT + nt * 16 + col0] = acc[nt][r];
    }
}

// ---------------------------------------------------------------------------
// Edge phase. Scores are relu(...)+1 >= 1 > 0, so positive-float bit patterns
// order like ints -> int atomicMax implements segment_max (m init = 0).
// ---------------------------------------------------------------------------
__global__ void edge_score_kernel(const int* __restrict__ src, const int* __restrict__ dst,
                                  const float* __restrict__ el, const float* __restrict__ er,
                                  float* __restrict__ e, int* __restrict__ m) {
    int i = blockIdx.x * 256 + threadIdx.x;
    if (i >= N_EDGES) return;
    float v = fmaxf(el[src[i]] + er[dst[i]], 0.f) + 1.f;
    e[i] = v;
    atomicMax(m + dst[i], __float_as_int(v));
}

__global__ void edge_exp_kernel(const int* __restrict__ dst, const int* __restrict__ m,
                                float* __restrict__ e, float* __restrict__ denom) {
    int i = blockIdx.x * 256 + threadIdx.x;
    if (i >= N_EDGES) return;
    int d = dst[i];
    float a = __expf(e[i] - __int_as_float(m[d]));
    e[i] = a;                                  // reuse buffer for numerator
    atomicAdd(denom + d, a);
}

// One wave per edge: gather ft[src], scale, atomic scatter-add into rst[dst].
// ft and rst are both L2-resident (51 MB each vs 192 MB L2).
__global__ __launch_bounds__(256)
void edge_agg_kernel(const int* __restrict__ src, const int* __restrict__ dst,
                     const float* __restrict__ a, const float* __restrict__ denom,
                     const float* __restrict__ ft, float* __restrict__ rst) {
    int lane = threadIdx.x & 31;
    int i    = blockIdx.x * 8 + (threadIdx.x >> 5);
    if (i >= N_EDGES) return;
    int s = src[i], d = dst[i];
    float coeff = a[i] / denom[d];
    const float* f = ft  + (size_t)s * DOUT;
    float*       r = rst + (size_t)d * DOUT;
    #pragma unroll
    for (int j = 0; j < 8; ++j)
        atomicAdd(r + j * 32 + lane, f[j * 32 + lane] * coeff);
}

// ---------------------------------------------------------------------------
extern "C" void kernel_launch(void* const* d_in, const int* in_sizes, int n_in,
                              void* d_out, int out_size, void* d_ws, size_t ws_size,
                              hipStream_t stream) {
    const float* feat = (const float*)d_in[0];
    // d_in[1] = p : unused in the eval-mode reference path
    const float* Wl   = (const float*)d_in[2];
    const float* bl   = (const float*)d_in[3];
    const float* Wr   = (const float*)d_in[4];
    const float* br   = (const float*)d_in[5];
    const float* W    = (const float*)d_in[6];
    const float* b    = (const float*)d_in[7];
    const int*   src  = (const int*)d_in[8];
    const int*   dst  = (const int*)d_in[9];
    float* rst = (float*)d_out;

    char* ws = (char*)d_ws;
    float*    ft    = (float*)ws;    ws += (size_t)N_NODES * DOUT * sizeof(float);
    float*    el    = (float*)ws;    ws += N_NODES * sizeof(float);
    float*    er    = (float*)ws;    ws += N_NODES * sizeof(float);
    float*    e     = (float*)ws;    ws += N_EDGES * sizeof(float);
    float*    m     = (float*)ws;    ws += N_NODES * sizeof(float);
    float*    denom = (float*)ws;    ws += N_NODES * sizeof(float);
    uint32_t* Wbf   = (uint32_t*)ws; ws += 65536 * sizeof(uint32_t);

    hipMemsetAsync(m,     0, N_NODES * sizeof(float), stream);
    hipMemsetAsync(denom, 0, N_NODES * sizeof(float), stream);
    hipMemsetAsync(d_out, 0, (size_t)N_NODES * DOUT * sizeof(float), stream);

    pack_w_kernel   <<<256, 256, 0, stream>>>(W, Wbf);
    node_attn_kernel<<<(N_NODES + 7) / 8, 256, 0, stream>>>(feat, Wl, Wr, bl, br, el, er);
    gemm_ft_kernel  <<<(STRIPES + 7) / 8, 256, 0, stream>>>(feat, Wbf, b, ft);
    edge_score_kernel<<<(N_EDGES + 255) / 256, 256, 0, stream>>>(src, dst, el, er, e, (int*)m);
    edge_exp_kernel  <<<(N_EDGES + 255) / 256, 256, 0, stream>>>(dst, (const int*)m, e, denom);
    edge_agg_kernel  <<<N_EDGES / 8, 256, 0, stream>>>(src, dst, e, denom, ft, rst);
}
